// FastHyperbolicMultiheadCrossAttention_5952824672946
// MI455X (gfx1250) — matrix operand order repaired and checked by reading, not compiled, not run on hardware
//
#include <hip/hip_runtime.h>
#include <hip/hip_bf16.h>
#include <stdint.h>

typedef __attribute__((ext_vector_type(16))) __bf16 v16bf;
typedef __attribute__((ext_vector_type(8)))  __bf16 v8bf;
typedef __attribute__((ext_vector_type(8)))  float  v8f;

#define EPS      1e-5f
#define MIN_NORM 1e-7f

// ---------------------------------------------------------------------------
// WMMA fragment loaders (CDNA5 layouts, wave32; CDNA5 ISA 7.12.2).
// Tiles staged so every lane's fragment is contiguous -> ds_load_b128.
// ---------------------------------------------------------------------------
__device__ __forceinline__ v16bf concat8(v8bf lo, v8bf hi) {
    return __builtin_shufflevector(lo, hi, 0, 1, 2, 3, 4, 5, 6, 7,
                                           8, 9, 10, 11, 12, 13, 14, 15);
}

// A fragment from row-major tile t[row][k] (ld multiple of 8 elements).
__device__ __forceinline__ v16bf load_a_frag(const __bf16* t, int ld, int row0, int k0, int lane) {
    const int m = lane & 15, half = lane >> 4;
    const __bf16* base = t + (row0 + m) * ld + k0 + half * 8;
    const v8bf lo = *(const v8bf*)(base);
    const v8bf hi = *(const v8bf*)(base + 16);
    return concat8(lo, hi);
}

// B fragment from tile stored as t[n][k] (k contiguous, ld multiple of 8).
__device__ __forceinline__ v16bf load_b_frag_nk(const __bf16* t, int ld, int n0, int k0, int lane) {
    const int n = lane & 15, half = lane >> 4;
    const __bf16* base = t + (n0 + n) * ld + k0 + half * 8;
    const v8bf lo = *(const v8bf*)(base);
    const v8bf hi = *(const v8bf*)(base + 16);
    return concat8(lo, hi);
}

// gfx1250 async global->LDS DMA (ASYNCcnt-tracked; CDNA5 ISA)
__device__ __forceinline__ void async_lds_b128(uint32_t lds_off, const __bf16* g) {
    asm volatile("global_load_async_to_lds_b128 %0, %1, off"
                 :: "v"(lds_off), "v"((unsigned long long)(uintptr_t)g)
                 : "memory");
}
__device__ __forceinline__ void wait_async0() {
    asm volatile("s_wait_asynccnt 0x0" ::: "memory");
}

// ---------------------------------------------------------------------------
// Kernel 0: fp32 -> bf16 bulk convert (one pass; GEMM tiles then stream as-is)
// ---------------------------------------------------------------------------
__global__ __launch_bounds__(256) void cvt_f32_to_bf16(const float* __restrict__ in,
                                                       __bf16* __restrict__ out, int n) {
    const int i = (blockIdx.x * 256 + threadIdx.x) * 4;
    if (i + 3 < n) {
        const float4 f = *(const float4*)(in + i);
        out[i + 0] = (__bf16)f.x; out[i + 1] = (__bf16)f.y;
        out[i + 2] = (__bf16)f.z; out[i + 3] = (__bf16)f.w;
    }
}

// ---------------------------------------------------------------------------
// Kernel 1: MX = X @ W^T  (X: MxK bf16, W: NCxK bf16, D fp32)
// 128 threads (4 waves), block tile 64x64, double-buffered async-LDS pipeline.
// ---------------------------------------------------------------------------
__device__ __forceinline__ void issue_tile(uint32_t xsb, uint32_t wsb,
                                           const __bf16* X, const __bf16* W,
                                           int m0, int n0, int K, int kc, int tid) {
    // 64 rows x 32 k bf16 per tile = 256 16B chunks; 2 per thread per tile
#pragma unroll
    for (int cc = 0; cc < 2; ++cc) {
        const int c   = tid + cc * 128;
        const int row = c >> 2, sub = c & 3;
        const uint32_t loff = (uint32_t)((row * 40 + sub * 8) * 2);
        async_lds_b128(xsb + loff, X + (size_t)(m0 + row) * K + kc + sub * 8);
        async_lds_b128(wsb + loff, W + (size_t)(n0 + row) * K + kc + sub * 8);
    }
}

__global__ __launch_bounds__(128) void gemm_xwt_bf16(const __bf16* __restrict__ X,
                                                     const __bf16* __restrict__ W,
                                                     float* __restrict__ D,
                                                     int M, int NC, int K) {
    __shared__ __bf16 Xs[2][64][40];
    __shared__ __bf16 Ws[2][64][40];

    const int tid  = threadIdx.x;
    const int lane = tid & 31;
    const int wid  = tid >> 5;
    const int n0   = blockIdx.x * 64;
    const int m0   = blockIdx.y * 64;

    const uint32_t xsb[2] = { (uint32_t)(uintptr_t)&Xs[0][0][0], (uint32_t)(uintptr_t)&Xs[1][0][0] };
    const uint32_t wsb[2] = { (uint32_t)(uintptr_t)&Ws[0][0][0], (uint32_t)(uintptr_t)&Ws[1][0][0] };

    v8f acc[4];
#pragma unroll
    for (int nt = 0; nt < 4; ++nt) acc[nt] = v8f{0.f,0.f,0.f,0.f,0.f,0.f,0.f,0.f};

    const int niter = K / 32;
    issue_tile(xsb[0], wsb[0], X, W, m0, n0, K, 0, tid);
    wait_async0();
    __syncthreads();

    for (int it = 0; it < niter; ++it) {
        if (it + 1 < niter)                                  // prefetch next tile via async DMA
            issue_tile(xsb[(it + 1) & 1], wsb[(it + 1) & 1], X, W, m0, n0, K, (it + 1) * 32, tid);

        const __bf16* xt = &Xs[it & 1][0][0];
        const __bf16* wt = &Ws[it & 1][0][0];
        const v16bf a = load_a_frag(xt, 40, wid * 16, 0, lane);
#pragma unroll
        for (int nt = 0; nt < 4; ++nt) {
            const v16bf bb = load_b_frag_nk(wt, 40, nt * 16, 0, lane);
            acc[nt] = __builtin_amdgcn_wmma_f32_16x16x32_bf16(false, a, false, bb,
                                                              (short)0, acc[nt], false, false);
        }
        wait_async0();                                       // own DMA for tile it+1 done
        __syncthreads();                                     // all waves ready / reads retired
    }

    const int half = lane >> 4, nl = lane & 15;
#pragma unroll
    for (int nt = 0; nt < 4; ++nt)
#pragma unroll
        for (int v = 0; v < 8; ++v) {
            const int row = m0 + wid * 16 + v + half * 8;
            const int col = n0 + nt * 16 + nl;
            D[(size_t)row * NC + col] = acc[nt][v];
        }
}

// ---------------------------------------------------------------------------
// Kernel 2: per-row mobius rescale + mobius_add(bias). OutT = float or bf16.
// ---------------------------------------------------------------------------
__device__ __forceinline__ float block_reduce_sum(float v, float* s) {
#pragma unroll
    for (int o = 16; o > 0; o >>= 1) v += __shfl_xor(v, o, 32);
    const int lane = threadIdx.x & 31, wid = threadIdx.x >> 5;
    __syncthreads();
    if (lane == 0) s[wid] = v;
    __syncthreads();
    float r = 0.f;
#pragma unroll
    for (int w = 0; w < 8; ++w) r += s[w];
    return r;
}

template <typename OutT>
__global__ __launch_bounds__(256) void hyp_rowfix(const float* __restrict__ X,
                                                  const float* __restrict__ MX,
                                                  const float* __restrict__ bvec,
                                                  OutT* __restrict__ OUT, int C) {
    __shared__ float red[8];
    const int row = blockIdx.x;
    const int tid = threadIdx.x;
    float mxv[4], bv[4];
    float x2 = 0.f, mx2 = 0.f, b2 = 0.f, mxb = 0.f;
#pragma unroll
    for (int i = 0; i < 4; ++i) {
        const int c  = tid + i * 256;
        const float xv = X[(size_t)row * C + c];
        const float mv = MX[(size_t)row * C + c];
        const float bb = bvec[c];
        x2 += xv * xv; mx2 += mv * mv; b2 += bb * bb; mxb += mv * bb;
        mxv[i] = mv; bv[i] = bb;
    }
    x2  = block_reduce_sum(x2,  red);
    mx2 = block_reduce_sum(mx2, red);
    b2  = block_reduce_sum(b2,  red);
    mxb = block_reduce_sum(mxb, red);

    const float xn  = fmaxf(sqrtf(x2),  MIN_NORM);
    const float mxn = fmaxf(sqrtf(mx2), MIN_NORM);
    const float xc  = fminf(xn, 1.f - 1e-7f);
    const float art = 0.5f * (log1pf(xc) - log1pf(-xc));            // artanh
    const float s   = tanhf(mxn / xn * art) / mxn;                  // mobius_matvec scale
    const float xy  = s * mxb;
    const float sm2 = s * s * mx2;
    const float c1  = 1.f + 2.f * xy + b2;
    const float c2  = 1.f - sm2;
    const float inv = 1.f / fmaxf(1.f + 2.f * xy + sm2 * b2, MIN_NORM);
#pragma unroll
    for (int i = 0; i < 4; ++i) {
        const int c = tid + i * 256;
        OUT[(size_t)row * C + c] = (OutT)((c1 * s * mxv[i] + c2 * bv[i]) * inv);
    }
}

// ---------------------------------------------------------------------------
// Kernel 3: flash-style hyperbolic attention over bf16 q/k/v.
// grid (N/64, H, B), block 128 (4 waves). Dh = 64.
// Emits fp32 result (for final rowfix x-norm) and bf16 result (for Wo GEMM).
// ---------------------------------------------------------------------------
__global__ __launch_bounds__(128) void hyp_flash_attn(const __bf16* __restrict__ Qp,
                                                      const __bf16* __restrict__ Kp,
                                                      const __bf16* __restrict__ Vp,
                                                      float* __restrict__ Op,
                                                      __bf16* __restrict__ Opb,
                                                      int N, int C) {
    const int q0 = blockIdx.x * 64;
    const int h  = blockIdx.y;
    const int b  = blockIdx.z;
    const int tid = threadIdx.x, lane = tid & 31, wid = tid >> 5;
    const int Dh = 64;

    __shared__ __bf16 Qs[64][72];   // [q][d]    (A of QK^T)
    __shared__ __bf16 Ks[64][72];   // [kk][d]   (B of QK^T, [n][k])
    __shared__ __bf16 Vs[64][72];   // [d][kk]   (B of P@V,  [n][k])
    __shared__ float  Sf[64][66];
    __shared__ __bf16 Pb[64][72];
    __shared__ float  q2s[64], k2s[64], mrow[64], lrow[64], crow[64];
    __shared__ float  part[64][2];

    const size_t hb = (size_t)b * N * C + (size_t)h * Dh;

    { // load Q block (bf16 copy), accumulate q^2
        const int r = tid >> 1, d0 = (tid & 1) * 32;
        const __bf16* p = Qp + hb + (size_t)(q0 + r) * C + d0;
        float acc = 0.f;
#pragma unroll
        for (int c8 = 0; c8 < 4; ++c8) {
            const v8bf v = *(const v8bf*)(p + c8 * 8);
            *(v8bf*)&Qs[r][d0 + c8 * 8] = v;
#pragma unroll
            for (int i = 0; i < 8; ++i) { const float f = (float)v[i]; acc += f * f; }
        }
        part[r][tid & 1] = acc;
    }
    if (tid < 64) { mrow[tid] = -1e30f; lrow[tid] = 0.f; }
    __syncthreads();
    if (tid < 64) q2s[tid] = fminf(part[tid][0] + part[tid][1], 1.f - EPS);

    v8f oacc[4];
#pragma unroll
    for (int dt = 0; dt < 4; ++dt) oacc[dt] = v8f{0.f,0.f,0.f,0.f,0.f,0.f,0.f,0.f};

    const int r0 = wid * 16;
    const int half = lane >> 4, nl = lane & 15;

    for (int j = 0; j < N / 64; ++j) {
        __syncthreads();
        { // K native copy + k^2; V transposed stage
            const int kk = tid >> 1, d0 = (tid & 1) * 32;
            const __bf16* pk = Kp + hb + (size_t)(j * 64 + kk) * C + d0;
            const __bf16* pv = Vp + hb + (size_t)(j * 64 + kk) * C + d0;
            float acc = 0.f;
#pragma unroll
            for (int c8 = 0; c8 < 4; ++c8) {
                const v8bf vk = *(const v8bf*)(pk + c8 * 8);
                const v8bf vv = *(const v8bf*)(pv + c8 * 8);
                *(v8bf*)&Ks[kk][d0 + c8 * 8] = vk;
#pragma unroll
                for (int i = 0; i < 8; ++i) {
                    const float f = (float)vk[i]; acc += f * f;
                    Vs[d0 + c8 * 8 + i][kk] = vv[i];
                }
            }
            part[kk][tid & 1] = acc;
        }
        __syncthreads();
        if (tid < 64) k2s[tid] = fminf(part[tid][0] + part[tid][1], 1.f - EPS);
        __syncthreads();

        // ---- S = Q @ K^T ----
        v8f sfr[4];
#pragma unroll
        for (int nt = 0; nt < 4; ++nt) sfr[nt] = v8f{0.f,0.f,0.f,0.f,0.f,0.f,0.f,0.f};
        const v16bf qa0 = load_a_frag(&Qs[0][0], 72, r0, 0,  lane);
        const v16bf qa1 = load_a_frag(&Qs[0][0], 72, r0, 32, lane);
#pragma unroll
        for (int nt = 0; nt < 4; ++nt) {
            const v16bf kb0 = load_b_frag_nk(&Ks[0][0], 72, nt * 16, 0,  lane);
            const v16bf kb1 = load_b_frag_nk(&Ks[0][0], 72, nt * 16, 32, lane);
            sfr[nt] = __builtin_amdgcn_wmma_f32_16x16x32_bf16(false, qa0, false, kb0, (short)0, sfr[nt], false, false);
            sfr[nt] = __builtin_amdgcn_wmma_f32_16x16x32_bf16(false, qa1, false, kb1, (short)0, sfr[nt], false, false);
        }

        // ---- hyperbolic distance scores ----
#pragma unroll
        for (int nt = 0; nt < 4; ++nt)
#pragma unroll
            for (int v = 0; v < 8; ++v) {
                const int row = r0 + v + half * 8;
                const int col = nt * 16 + nl;
                const float qk = sfr[nt][v];
                const float q2 = q2s[row], k2 = k2s[col];
                const float den   = (1.f - q2) * (1.f - k2) + EPS;
                const float delta = 2.f * (qk - q2 * k2) / den;
                const float dist  = sqrtf(fmaxf(delta, EPS));
                Sf[row][col] = -dist * 0.125f;
            }
        __syncthreads();

        // ---- online softmax ----
        { const int r = tid >> 1, sg = tid & 1;
          float pm = -1e30f;
#pragma unroll
          for (int c = 0; c < 32; ++c) pm = fmaxf(pm, Sf[r][sg * 32 + c]);
          part[r][sg] = pm; }
        __syncthreads();
        if ((tid & 1) == 0) {
            const int r = tid >> 1;
            const float mj = fmaxf(part[r][0], part[r][1]);
            const float mn = fmaxf(mrow[r], mj);
            crow[r] = __expf(mrow[r] - mn);
            mrow[r] = mn;
        }
        __syncthreads();
#pragma unroll
        for (int dt = 0; dt < 4; ++dt)
#pragma unroll
            for (int v = 0; v < 8; ++v) oacc[dt][v] *= crow[r0 + v + half * 8];
        { const int r = tid >> 1, sg = tid & 1;
          const float m = mrow[r]; float ps = 0.f;
#pragma unroll
          for (int c = 0; c < 32; ++c) {
              const float p = __expf(Sf[r][sg * 32 + c] - m);
              Pb[r][sg * 32 + c] = (__bf16)p;
              ps += p;
          }
          part[r][sg] = ps; }
        __syncthreads();
        if ((tid & 1) == 0) {
            const int r = tid >> 1;
            lrow[r] = lrow[r] * crow[r] + part[r][0] + part[r][1];
        }
        __syncthreads();

        // ---- O += P @ V ----
        const v16bf pa0 = load_a_frag(&Pb[0][0], 72, r0, 0,  lane);
        const v16bf pa1 = load_a_frag(&Pb[0][0], 72, r0, 32, lane);
#pragma unroll
        for (int dt = 0; dt < 4; ++dt) {
            const v16bf vb0 = load_b_frag_nk(&Vs[0][0], 72, dt * 16, 0,  lane);
            const v16bf vb1 = load_b_frag_nk(&Vs[0][0], 72, dt * 16, 32, lane);
            oacc[dt] = __builtin_amdgcn_wmma_f32_16x16x32_bf16(false, pa0, false, vb0, (short)0, oacc[dt], false, false);
            oacc[dt] = __builtin_amdgcn_wmma_f32_16x16x32_bf16(false, pa1, false, vb1, (short)0, oacc[dt], false, false);
        }
    }
    __syncthreads();

#pragma unroll
    for (int dt = 0; dt < 4; ++dt)
#pragma unroll
        for (int v = 0; v < 8; ++v) {
            const int row = r0 + v + half * 8;
            const int col = dt * 16 + nl;
            const float val = oacc[dt][v] / lrow[row];
            const size_t o  = hb + (size_t)(q0 + row) * C + col;
            Op[o]  = val;
            Opb[o] = (__bf16)val;
        }
}

// ---------------------------------------------------------------------------
// Host launcher
// ---------------------------------------------------------------------------
extern "C" void kernel_launch(void* const* d_in, const int* in_sizes, int n_in,
                              void* d_out, int out_size, void* d_ws, size_t ws_size,
                              hipStream_t stream) {
    const float* query = (const float*)d_in[0];
    const float* key   = (const float*)d_in[1];
    const float* value = (const float*)d_in[2];
    const float* Wq = (const float*)d_in[3]; const float* bq = (const float*)d_in[4];
    const float* Wk = (const float*)d_in[5]; const float* bk = (const float*)d_in[6];
    const float* Wv = (const float*)d_in[7]; const float* bv = (const float*)d_in[8];
    const float* Wo = (const float*)d_in[9]; const float* bo = (const float*)d_in[10];

    const int C = 1024;
    const int M = in_sizes[0] / C;     // B*N = 4096
    const int B = 4;
    const int N = M / B;               // 1024
    const int H = 16;

    const size_t tsz = (size_t)M * C;
    const size_t wsz = (size_t)C * C;
    // mx(f32) + attn(f32) + xbf + wbf + qpb + kpb + vpb (bf16)
    const size_t need = tsz * 4 * 2 + (tsz * 4 + wsz) * 2;
    if (ws_size < need) return;

    float*  mx   = (float*)d_ws;
    float*  attn = mx + tsz;
    __bf16* xbf  = (__bf16*)(attn + tsz);
    __bf16* wbf  = xbf + tsz;
    __bf16* qpb  = wbf + wsz;
    __bf16* kpb  = qpb + tsz;
    __bf16* vpb  = kpb + tsz;

    const dim3 gGemm(C / 64, M / 64);
    const dim3 bGemm(128);
    const int  gCvtX = (int)(tsz / 1024);
    const int  gCvtW = (int)(wsz / 1024);

    // q projection
    cvt_f32_to_bf16<<<gCvtX, 256, 0, stream>>>(query, xbf, (int)tsz);
    cvt_f32_to_bf16<<<gCvtW, 256, 0, stream>>>(Wq, wbf, (int)wsz);
    gemm_xwt_bf16<<<gGemm, bGemm, 0, stream>>>(xbf, wbf, mx, M, C, C);
    hyp_rowfix<__bf16><<<M, 256, 0, stream>>>(query, mx, bq, qpb, C);
    // k projection
    cvt_f32_to_bf16<<<gCvtX, 256, 0, stream>>>(key, xbf, (int)tsz);
    cvt_f32_to_bf16<<<gCvtW, 256, 0, stream>>>(Wk, wbf, (int)wsz);
    gemm_xwt_bf16<<<gGemm, bGemm, 0, stream>>>(xbf, wbf, mx, M, C, C);
    hyp_rowfix<__bf16><<<M, 256, 0, stream>>>(key, mx, bk, kpb, C);
    // v projection
    cvt_f32_to_bf16<<<gCvtX, 256, 0, stream>>>(value, xbf, (int)tsz);
    cvt_f32_to_bf16<<<gCvtW, 256, 0, stream>>>(Wv, wbf, (int)wsz);
    gemm_xwt_bf16<<<gGemm, bGemm, 0, stream>>>(xbf, wbf, mx, M, C, C);
    hyp_rowfix<__bf16><<<M, 256, 0, stream>>>(value, mx, bv, vpb, C);

    // hyperbolic-distance flash attention (fp32 + bf16 outputs)
    const dim3 gAttn(N / 64, H, B);
    hyp_flash_attn<<<gAttn, dim3(128), 0, stream>>>(qpb, kpb, vpb, attn, xbf, N, C);

    // output projection
    cvt_f32_to_bf16<<<gCvtW, 256, 0, stream>>>(Wo, wbf, (int)wsz);
    gemm_xwt_bf16<<<gGemm, bGemm, 0, stream>>>(xbf, wbf, mx, M, C, C);
    hyp_rowfix<float><<<M, 256, 0, stream>>>(attn, mx, bo, (float*)d_out, C);
}
